// LSTM_88828513616451
// MI455X (gfx1250) — matrix operand order repaired and checked by reading, not compiled, not run on hardware
//
#include <hip/hip_runtime.h>

// ---------------------------------------------------------------------------
// LSTM on MI455X (gfx1250, wave32, WMMA 16x16x32 f16 with f32 accumulate)
//
// Phase A (full-GPU GEMM): Gx = x_chunk @ Wx + b for both layers, stored in
//   WMMA C-fragment layout so the recurrence initializes accumulators with a
//   single v8f load per tile. Double-buffered A/B fragments for load/WMMA
//   overlap.
// Phase B (persistent recurrence): 1 workgroup per layer (layers independent),
//   16 waves; wave w owns h-column tile w and computes its f/i/g/o gate tiles
//   for all 4 batch row-tiles (16 WMMA tiles, 128 wmma/step). h state lives in
//   LDS (f16, padded rows), c state lives in registers (exclusive per-lane
//   ownership). Wh streamed from L2 in B-fragment order, double-buffered;
//   next step's Gx prefetched (global_prefetch_b8) during the K-loop.
// ---------------------------------------------------------------------------

typedef _Float16 v16h __attribute__((ext_vector_type(16)));
typedef _Float16 v8h  __attribute__((ext_vector_type(8)));
typedef float    v8f  __attribute__((ext_vector_type(8)));
typedef float    v4f  __attribute__((ext_vector_type(4)));

#define T_TOTAL 2048
#define BATCH   64
#define DIN     256
#define HID     256
#define LAYERS  2
#define GATES   1024          // 4*H
#define TC      64            // timesteps per chunk
#define NCHUNK  (T_TOTAL / TC)
#define HPITCH  264           // LDS h16 row pitch in halves (528B, 16B aligned)

__device__ __forceinline__ float sigm_f(float x) {
    return 1.0f / (1.0f + __expf(-x));
}
__device__ __forceinline__ float tanh_f(float x) {
    return 2.0f / (1.0f + __expf(-2.0f * x)) - 1.0f;
}

// Pack W (fp32, [L, 512, 1024]) into WMMA B-fragment order, f16:
//   dst[((l*64 + nt)*8 + kc)*32 + lane][16 halves]
//   lane = (col%16) + 16*(((k%32)/8)&1), element e -> K = kc*32 + (lane<16?0:8) + (e<8 ? e : 16 + e - 8)
__global__ void pack_w_kernel(const float* __restrict__ W,
                              _Float16* __restrict__ Wxp,
                              _Float16* __restrict__ Whp) {
    int gid = blockIdx.x * blockDim.x + threadIdx.x;   // < 2*2*262144
    int m   = gid >> 19;            // 0 = Wx, 1 = Wh
    int rem = gid & 524287;
    int l   = rem >> 18;
    int e   = rem & 262143;
    int k   = e >> 10;              // 0..255
    int col = e & 1023;
    int nt  = col >> 4;
    int kc  = k >> 5;
    int lane = (col & 15) + (((k >> 3) & 1) << 4);
    int ele  = (k & 7) | (((k >> 4) & 1) << 3);
    float v = W[((size_t)(l * 512) + (m ? 256 + k : k)) * 1024 + col];
    _Float16* dst = m ? Whp : Wxp;
    dst[((((size_t)l * 64 + nt) * 8 + kc) * 32 + lane) * 16 + ele] = (_Float16)v;
}

// Initialize carried state from h0/c0 inputs.
__global__ void init_state_kernel(const float* __restrict__ h0,
                                  const float* __restrict__ c0,
                                  _Float16* __restrict__ h16s,
                                  float* __restrict__ cstate) {
    int i = blockIdx.x * blockDim.x + threadIdx.x;     // < L*B*H = 32768
    h16s[i]   = (_Float16)h0[i];
    cstate[i] = c0[i];
}

// Build an A fragment (16x32 f16 layout) from an fp32 row in global memory.
__device__ __forceinline__ v16h load_a_from_x(const float* xrow, int kb) {
    v4f a0 = *(const v4f*)(xrow + kb);
    v4f a1 = *(const v4f*)(xrow + kb + 4);
    v4f a2 = *(const v4f*)(xrow + kb + 16);
    v4f a3 = *(const v4f*)(xrow + kb + 20);
    v16h a;
#pragma unroll
    for (int i = 0; i < 4; ++i) {
        a[i]      = (_Float16)a0[i];
        a[4 + i]  = (_Float16)a1[i];
        a[8 + i]  = (_Float16)a2[i];
        a[12 + i] = (_Float16)a3[i];
    }
    return a;
}

// Phase A: Gx[l, tloc, r, nt, lane, 0..7] = (x_chunk @ Wx[l] + b[l]) fragments.
// grid = (32, 8, 2), block = 256 (8 waves). Wave handles 1 M-tile x 8 N-tiles.
__global__ void __launch_bounds__(256)
gemm_x_kernel(const float* __restrict__ x, const float* __restrict__ bias,
              const _Float16* __restrict__ Wxp, float* __restrict__ Gx, int t0) {
    const int lane = threadIdx.x & 31;
    const int wave = threadIdx.x >> 5;        // 0..7
    const int l    = blockIdx.z;
    const int nb   = blockIdx.y;              // 0..7
    const int m    = blockIdx.x * 8 + wave;   // M-tile 0..255 within chunk
    const int row  = t0 * BATCH + m * 16 + (lane & 15);
    const int k0   = (lane >> 4) << 3;        // 0 or 8 (A-fragment half select)
    const float* xrow = x + (size_t)row * DIN;
    const v16h* Bp = (const v16h*)Wxp + (size_t)l * 64 * 8 * 32;

    v8f acc[8] = {};

    // Software-pipelined: A and B fragments double-buffered so loads for the
    // next fragment overlap the current v_wmma instead of wait(loadcnt==0).
    v16h a_cur = load_a_from_x(xrow, k0);
    v16h b_cur = Bp[((size_t)(nb * 8) * 8 + 0) * 32 + lane];

#pragma unroll
    for (int kc = 0; kc < 8; ++kc) {
        v16h a_nxt = (kc < 7) ? load_a_from_x(xrow, (kc + 1) * 32 + k0) : a_cur;
#pragma unroll
        for (int j = 0; j < 8; ++j) {
            v16h b_nxt;
            if (j < 7)
                b_nxt = Bp[((size_t)(nb * 8 + j + 1) * 8 + kc) * 32 + lane];
            else if (kc < 7)
                b_nxt = Bp[((size_t)(nb * 8) * 8 + (kc + 1)) * 32 + lane];
            else
                b_nxt = b_cur;
            acc[j] = __builtin_amdgcn_wmma_f32_16x16x32_f16(
                false, a_cur, false, b_cur, (short)0, acc[j], false, false);
            b_cur = b_nxt;
        }
        a_cur = a_nxt;
    }

    const int tloc = m >> 2, r = m & 3;
#pragma unroll
    for (int j = 0; j < 8; ++j) {
        const int nt = nb * 8 + j;
        const float bb = bias[l * GATES + nt * 16 + (lane & 15)];
        v8f v = acc[j];
#pragma unroll
        for (int i = 0; i < 8; ++i) v[i] += bb;
        // Regular (RT) store: re-read by the next kernel; keep it in L2.
        *(v8f*)(Gx + ((((size_t)(l * TC + tloc) * 4 + r) * 64 + nt) * 32 + lane) * 8) = v;
    }
}

// Phase B: persistent per-layer recurrence over one chunk.
// grid = (LAYERS), block = 512 (16 waves). Wave ht owns h-column tile ht.
__global__ void __launch_bounds__(512)
lstm_recur_kernel(const float* __restrict__ Gx, const _Float16* __restrict__ Whp,
                  _Float16* __restrict__ h16s, float* __restrict__ cstate,
                  float* __restrict__ out, int t0, int last) {
    __shared__ __attribute__((aligned(16))) _Float16 hsm[64 * HPITCH];

    const int l    = blockIdx.x;
    const int tid  = threadIdx.x;
    const int lane = tid & 31;
    const int ht   = tid >> 5;                 // wave id = h-column tile 0..15
    const int k0   = (lane >> 4) << 3;         // A-fragment half select
    const int colq = lane & 15;
    const int col  = ht * 16 + colq;           // this lane's h column
    const int rhal = (lane >> 4) << 3;         // +8 rows for upper lane half (C layout)

    // Load carried h (f16) into LDS.
    for (int i = tid; i < BATCH * HID; i += 512) {
        hsm[(i >> 8) * HPITCH + (i & 255)] = h16s[(size_t)l * BATCH * HID + i];
    }
    // Load carried c into registers (exclusive per-lane ownership).
    float creg[4][8];
#pragma unroll
    for (int r = 0; r < 4; ++r)
#pragma unroll
        for (int j = 0; j < 8; ++j) {
            const int row = r * 16 + rhal + j;
            creg[r][j] = cstate[(size_t)l * BATCH * HID + row * HID + col];
        }
    __syncthreads();

    const v16h* Bp = (const v16h*)Whp + (size_t)l * 64 * 8 * 32;

    for (int ts = 0; ts < TC; ++ts) {
        // Accumulators initialized from precomputed x-part (fragment layout).
        // Touch-once data: non-temporal loads so Gx doesn't thrash L2/WGP$.
        v8f acc[4][4];
        const float* gbase = Gx + (size_t)(l * TC + ts) * 4 * 64 * 32 * 8;
#pragma unroll
        for (int r = 0; r < 4; ++r)
#pragma unroll
            for (int g = 0; g < 4; ++g)
                acc[r][g] = __builtin_nontemporal_load((const v8f*)(gbase +
                    (((size_t)r * 64 + (g * 16 + ht)) * 32 + lane) * 8));

        // Prefetch next step's Gx fragments (overlaps with this step's GEMM).
        if (ts + 1 < TC) {
            const float* gn = gbase + (size_t)4 * 64 * 32 * 8;
#pragma unroll
            for (int r = 0; r < 4; ++r)
#pragma unroll
                for (int g = 0; g < 4; ++g)
                    __builtin_prefetch(gn +
                        (((size_t)r * 64 + (g * 16 + ht)) * 32 + lane) * 8, 0, 0);
        }

        // Gate GEMM: gates += h_{t-1} @ Wh (K = 256 over 8 chunks of 32).
        // A (from LDS) and B (from L2) fragments double-buffered.
        v16h a_cur[4], a_nxt[4];
#pragma unroll
        for (int r = 0; r < 4; ++r) {
            const int rowb = r * 16 + (lane & 15);
            union { v16h v; v8h h[2]; } u;
            u.h[0] = *(const v8h*)&hsm[rowb * HPITCH + k0];
            u.h[1] = *(const v8h*)&hsm[rowb * HPITCH + k0 + 16];
            a_cur[r] = u.v;
        }
        v16h b_cur = Bp[((size_t)ht * 8 + 0) * 32 + lane];   // g=0, kc=0

#pragma unroll
        for (int kc = 0; kc < 8; ++kc) {
            if (kc < 7) {
                const int kb = (kc + 1) * 32 + k0;
#pragma unroll
                for (int r = 0; r < 4; ++r) {
                    const int rowb = r * 16 + (lane & 15);
                    union { v16h v; v8h h[2]; } u;
                    u.h[0] = *(const v8h*)&hsm[rowb * HPITCH + kb];
                    u.h[1] = *(const v8h*)&hsm[rowb * HPITCH + kb + 16];
                    a_nxt[r] = u.v;
                }
            }
#pragma unroll
            for (int g = 0; g < 4; ++g) {
                v16h b_nxt;
                if (g < 3)
                    b_nxt = Bp[((size_t)((g + 1) * 16 + ht) * 8 + kc) * 32 + lane];
                else if (kc < 7)
                    b_nxt = Bp[((size_t)ht * 8 + (kc + 1)) * 32 + lane];
                else
                    b_nxt = b_cur;
#pragma unroll
                for (int r = 0; r < 4; ++r)
                    acc[r][g] = __builtin_amdgcn_wmma_f32_16x16x32_f16(
                        false, a_cur[r], false, b_cur, (short)0, acc[r][g],
                        false, false);
                b_cur = b_nxt;
            }
            if (kc < 7) {
#pragma unroll
                for (int r = 0; r < 4; ++r) a_cur[r] = a_nxt[r];
            }
        }
        __syncthreads();   // all reads of h_{t-1} complete before overwrite

        // Elementwise gate math; update c (regs), h (LDS f16), outputs.
        const int t = t0 + ts;
#pragma unroll
        for (int r = 0; r < 4; ++r) {
#pragma unroll
            for (int j = 0; j < 8; ++j) {
                const int row = r * 16 + rhal + j;
                const float fg = sigm_f(acc[r][0][j]);
                const float ig = sigm_f(acc[r][1][j]);
                const float gg = tanh_f(acc[r][2][j]);
                const float og = sigm_f(acc[r][3][j]);
                const float cn = fg * creg[r][j] + ig * gg;
                const float hn = og * tanh_f(cn);
                creg[r][j] = cn;
                hsm[row * HPITCH + col] = (_Float16)hn;
                if (l == LAYERS - 1)
                    __builtin_nontemporal_store(
                        hn, &out[((size_t)t * BATCH + row) * HID + col]);
                if (last && ts == TC - 1) {
                    const size_t tail = (size_t)T_TOTAL * BATCH * HID;
                    out[tail + ((size_t)l * BATCH + row) * HID + col] = hn;
                    out[tail + (size_t)LAYERS * BATCH * HID +
                        ((size_t)l * BATCH + row) * HID + col] = cn;
                }
            }
        }
        __syncthreads();   // h_t visible before next step's GEMM reads
    }

    // Persist carried state for the next chunk.
    for (int i = tid; i < BATCH * HID; i += 512) {
        h16s[(size_t)l * BATCH * HID + i] = hsm[(i >> 8) * HPITCH + (i & 255)];
    }
#pragma unroll
    for (int r = 0; r < 4; ++r)
#pragma unroll
        for (int j = 0; j < 8; ++j) {
            const int row = r * 16 + rhal + j;
            cstate[(size_t)l * BATCH * HID + row * HID + col] = creg[r][j];
        }
}

extern "C" void kernel_launch(void* const* d_in, const int* in_sizes, int n_in,
                              void* d_out, int out_size, void* d_ws, size_t ws_size,
                              hipStream_t stream) {
    (void)in_sizes; (void)n_in; (void)out_size; (void)ws_size;
    const float* x  = (const float*)d_in[0];
    const float* h0 = (const float*)d_in[1];
    const float* c0 = (const float*)d_in[2];
    const float* W  = (const float*)d_in[3];
    const float* b  = (const float*)d_in[4];
    float* out = (float*)d_out;

    // Workspace layout (~36 MB total).
    char* ws = (char*)d_ws;
    size_t off = 0;
    _Float16* Wxp = (_Float16*)(ws + off);
    off += (size_t)LAYERS * 64 * 8 * 32 * 16 * sizeof(_Float16);   // 1 MB
    _Float16* Whp = (_Float16*)(ws + off);
    off += (size_t)LAYERS * 64 * 8 * 32 * 16 * sizeof(_Float16);   // 1 MB
    _Float16* h16s = (_Float16*)(ws + off);
    off += (size_t)LAYERS * BATCH * HID * sizeof(_Float16);        // 64 KB
    float* cstate = (float*)(ws + off);
    off += (size_t)LAYERS * BATCH * HID * sizeof(float);           // 128 KB
    off = (off + 255) & ~(size_t)255;
    float* Gx = (float*)(ws + off);                                 // 33.5 MB

    pack_w_kernel<<<4096, 256, 0, stream>>>(W, Wxp, Whp);
    init_state_kernel<<<128, 256, 0, stream>>>(h0, c0, h16s, cstate);

    for (int c = 0; c < NCHUNK; ++c) {
        gemm_x_kernel<<<dim3(32, 8, 2), 256, 0, stream>>>(x, b, Wxp, Gx, c * TC);
        lstm_recur_kernel<<<dim3(LAYERS), 512, 0, stream>>>(
            Gx, Whp, h16s, cstate, out, c * TC, (c == NCHUNK - 1) ? 1 : 0);
    }
}
